// CKConv_13950053777885
// MI455X (gfx1250) — matrix operand order repaired and checked by reading, not compiled, not run on hardware
//
#include <hip/hip_runtime.h>
#include <hip/hip_bf16.h>

// CKConv on gfx1250: SIREN kernel generation (tiny) + causal conv1d as an
// implicit Toeplitz GEMM on the bf16 WMMA pipe (v_wmma_f32_16x16x32_bf16),
// with Tensor-Data-Mover double-buffered LDS staging of the kernel taps.
// Compute-bound: ~550 GFLOP vs ~40MB L2-resident data.

#define B_    64
#define CIN_  32
#define COUT_ 32
#define L_    2048
#define K_    2049
#define KP_   2080           // taps padded to 65 chunks of 32 (zeros past 2049)
#define KB_   416            // taps per LDS stage block (13 chunks of 32)
#define NCB_  (KP_ / KB_)    // 5 stage blocks per ci
#define NIT_  (CIN_ * NCB_)  // 160 staging iterations
#define LT_   256            // l-tile per workgroup
#define XPN_  (LT_ + KP_)    // packed x-window length (2336)

typedef __bf16    v16bf __attribute__((ext_vector_type(16)));
typedef float     v8f   __attribute__((ext_vector_type(8)));
typedef float     v4f   __attribute__((ext_vector_type(4)));
typedef unsigned  v4u   __attribute__((ext_vector_type(4)));

union AFrag { unsigned u[8]; v16bf v; };   // Hankel x patches (16x32 bf16, A layout)
union BFrag { v4u q[2];      v16bf v; };   // kernel taps     (32x16 bf16, B layout)

static __device__ __forceinline__ unsigned short f2bf(float f) {
  union { float f; unsigned u; } c; c.f = f;
  unsigned u = c.u;
  u = u + 0x7FFFu + ((u >> 16) & 1u);       // round-to-nearest-even
  return (unsigned short)(u >> 16);
}

// ---------------- TDM staging (gfx1250 Tensor Data Mover) --------------------
#if __has_builtin(__builtin_amdgcn_tensor_load_to_lds) && \
    __has_builtin(__builtin_amdgcn_s_wait_tensorcnt)
#define HAVE_TDM 1
typedef unsigned u32x4 __attribute__((ext_vector_type(4)));
typedef int      i32x4 __attribute__((ext_vector_type(4)));
typedef int      i32x8 __attribute__((ext_vector_type(8)));

// 2D tile load: 416 x 32 bf16 elements from a tensor with dim0 stride 2080,
// into a contiguous [32][416] LDS buffer. D# per CDNA5 ISA ch.8 tables.
static __device__ __forceinline__ void tdm_load_tile(const unsigned short* gsrc,
                                                     unsigned lds_addr) {
  const unsigned long long ga = (unsigned long long)(uintptr_t)gsrc;
  u32x4 g0;
  g0[0] = 1u;                                            // count=1 (valid), user mode
  g0[1] = lds_addr;                                      // LDS byte address
  g0[2] = (unsigned)(ga & 0xFFFFFFFFu);                  // global_addr[31:0]
  g0[3] = (unsigned)((ga >> 32) & 0x01FFFFFFu) | (2u << 30);  // addr[56:32] | type=2
  i32x8 g1;
  g1[0] = (int)(1u << 16);                               // data_size=1 (2B), no mcast
  g1[1] = (int)(((unsigned)KP_ & 0xFFFFu) << 16);        // tensor_dim0[15:0] @ [63:48]
  g1[2] = (int)((((unsigned)KP_ >> 16) & 0xFFFFu) | (32u << 16)); // dim0 hi | tensor_dim1 lo
  g1[3] = (int)((unsigned)KB_ << 16);                    // tile_dim0=416 @ [127:112]
  g1[4] = 32;                                            // tile_dim1=32 rows
  g1[5] = KP_;                                           // tensor_dim0_stride[31:0]
  g1[6] = 0;
  g1[7] = 0;
  i32x4 gz = {0, 0, 0, 0};
#if defined(__clang_major__) && (__clang_major__ >= 23)
  i32x8 gz8 = {0, 0, 0, 0, 0, 0, 0, 0};
  __builtin_amdgcn_tensor_load_to_lds(g0, g1, gz, gz, gz8, 0);
#else
  __builtin_amdgcn_tensor_load_to_lds(g0, g1, gz, gz, 0);
#endif
}
#endif

// ---------------- Phase 1: SIREN kernel net -> bf16 taps [ci][co][KP_] -------
__global__ void ck_siren(const float* __restrict__ w1, const float* __restrict__ b1,
                         const float* __restrict__ om1p,
                         const float* __restrict__ w2, const float* __restrict__ b2,
                         const float* __restrict__ om2p,
                         const float* __restrict__ w3, const float* __restrict__ b3,
                         unsigned short* __restrict__ kbf) {
  const int k = blockIdx.x;       // 0..KP_-1
  const int h = threadIdx.x;      // 32 threads (one wave)
  if (k >= K_) {                  // zero tail taps (k = 2049..2079), ci == h
    for (int co = 0; co < COUT_; ++co) kbf[(size_t)(h * COUT_ + co) * KP_ + k] = 0;
    return;
  }
  __shared__ float h1s[32], h2s[32];
  const float t   = -1.0f + (float)k * (1.0f / 1024.0f);   // linspace(-1,1,2049)
  const float om1 = om1p[0], om2 = om2p[0];
  h1s[h] = sinf(om1 * (w1[h] * t + b1[h]));
  __syncthreads();
  float s = b2[h];
#pragma unroll
  for (int j = 0; j < 32; ++j) s += w2[h * 32 + j] * h1s[j];
  h2s[h] = sinf(om2 * s);
  __syncthreads();
  for (int i = 0; i < 32; ++i) {
    const int o = i * 32 + h;                 // o = co*CIN + ci
    float s3 = b3[o];
#pragma unroll
    for (int j = 0; j < 32; ++j) s3 += w3[o * 32 + j] * h2s[j];
    const int co = o >> 5, ci = o & 31;
    kbf[(size_t)(ci * COUT_ + co) * KP_ + k] = f2bf(s3);
  }
}

// ---------------- Phase 2: causal conv as implicit GEMM on WMMA --------------
// Workgroup: 256 threads = 8 waves. b = blockIdx.y, l-tile of 256 = blockIdx.x.
// Wave w owns l-subtiles (2w, 2w+1) x co-tiles (0,1): 4 accumulators.
// Kernel-tap tiles are double-buffered in LDS; wave 0 drives the TDM.
__launch_bounds__(256)
__global__ void ck_conv(const float* __restrict__ x,
                        const unsigned short* __restrict__ kbf,
                        const float* __restrict__ bias,
                        float* __restrict__ out) {
  __shared__ alignas(16) unsigned short ks[2][COUT_ * KB_];  // 2 x [co][KB_] tiles
  __shared__ unsigned xp[XPN_];  // packed bf16 pairs (x[j], x[j+1]), j = l0+idx-2048

  const int tid  = threadIdx.x;
  const int w    = tid >> 5;
  const int lane = tid & 31;
  const int half = lane >> 4;
  const int lr   = lane & 15;
  const int l0   = blockIdx.x * LT_;
  const int b    = blockIdx.y;
  const bool issuer = (tid < 32);   // wave 0 drives the TDM

  v8f acc00 = {}, acc01 = {}, acc10 = {}, acc11 = {};

#ifdef HAVE_TDM
  if (issuer) tdm_load_tile(kbf, (unsigned)(uintptr_t)&ks[0][0]);
#endif

  for (int it = 0; it < NIT_; ++it) {
    const int ci  = it / NCB_;
    const int cb  = it - ci * NCB_;
    const int cur = it & 1;

#ifdef HAVE_TDM
    if (issuer) __builtin_amdgcn_s_wait_tensorcnt(0);   // tile `it` landed
    __syncthreads();                                    // everyone sees ks[cur];
                                                        // readers of ks[1-cur] done
    {
      const int nit = it + 1;
      if (issuer && nit < NIT_) {
        const int nci = nit / NCB_;
        const int ncb = nit - nci * NCB_;
        tdm_load_tile(kbf + ((size_t)(nci * COUT_)) * KP_ + ncb * KB_,
                      (unsigned)(uintptr_t)&ks[nit & 1][0]);  // overlaps compute
      }
    }
    if (cb == 0) {   // new ci: restage the packed x window, then publish it
      const float* xr = x + ((size_t)(b * CIN_ + ci)) * L_;
      for (int idx = tid; idx < XPN_; idx += 256) {
        const int j  = l0 + idx - (K_ - 1);
        const float f0 = (j     >= 0 && j     < L_) ? xr[j]     : 0.0f;
        const float f1 = (j + 1 >= 0 && j + 1 < L_) ? xr[j + 1] : 0.0f;
        xp[idx] = ((unsigned)f2bf(f1) << 16) | (unsigned)f2bf(f0);
      }
      __syncthreads();
    }
#else
    __syncthreads();                                    // readers of ks[cur] done
    if (cb == 0) {
      const float* xr = x + ((size_t)(b * CIN_ + ci)) * L_;
      for (int idx = tid; idx < XPN_; idx += 256) {
        const int j  = l0 + idx - (K_ - 1);
        const float f0 = (j     >= 0 && j     < L_) ? xr[j]     : 0.0f;
        const float f1 = (j + 1 >= 0 && j + 1 < L_) ? xr[j + 1] : 0.0f;
        xp[idx] = ((unsigned)f2bf(f1) << 16) | (unsigned)f2bf(f0);
      }
    }
    for (int idx = tid; idx < COUT_ * (KB_ / 8); idx += 256) {  // 1664 x 16B
      const int co = idx / (KB_ / 8);
      const int t_ = idx - co * (KB_ / 8);
      ((v4u*)ks[cur])[co * (KB_ / 8) + t_] =
          *(const v4u*)(kbf + ((size_t)(ci * COUT_ + co)) * KP_ + cb * KB_ + t_ * 8);
    }
    __syncthreads();
#endif

    const unsigned short* kst = ks[cur];
    for (int kc = 0; kc < KB_ / 32; ++kc) {   // 13 K-chunks of 32
      const int k0l = kc * 32;
      const int k0g = cb * KB_ + k0l;
      // B fragments: kernel tile, lane = co_local, K pairs contiguous -> 2x b128
      BFrag bf0, bf1;
      {
        const v4u* p0 = (const v4u*)&kst[lr * KB_ + k0l + 8 * half];
        bf0.q[0] = p0[0]; bf0.q[1] = p0[2];
        const v4u* p1 = (const v4u*)&kst[(16 + lr) * KB_ + k0l + 8 * half];
        bf1.q[0] = p1[0]; bf1.q[1] = p1[2];
      }
      // A fragments: Hankel x patches, pre-packed pairs -> aligned b32 (2addr) each
      AFrag af0, af1;
      {
        int base = (w * 2) * 16 + lr + k0g + 8 * half;
        af0.u[0] = xp[base + 0];  af0.u[1] = xp[base + 2];
        af0.u[2] = xp[base + 4];  af0.u[3] = xp[base + 6];
        af0.u[4] = xp[base + 16]; af0.u[5] = xp[base + 18];
        af0.u[6] = xp[base + 20]; af0.u[7] = xp[base + 22];
        base += 16;
        af1.u[0] = xp[base + 0];  af1.u[1] = xp[base + 2];
        af1.u[2] = xp[base + 4];  af1.u[3] = xp[base + 6];
        af1.u[4] = xp[base + 16]; af1.u[5] = xp[base + 18];
        af1.u[6] = xp[base + 20]; af1.u[7] = xp[base + 22];
      }
      acc00 = __builtin_amdgcn_wmma_f32_16x16x32_bf16(false, af0.v, false, bf0.v,
                                                      (short)0, acc00, false, false);
      acc01 = __builtin_amdgcn_wmma_f32_16x16x32_bf16(false, af0.v, false, bf1.v,
                                                      (short)0, acc01, false, false);
      acc10 = __builtin_amdgcn_wmma_f32_16x16x32_bf16(false, af1.v, false, bf0.v,
                                                      (short)0, acc10, false, false);
      acc11 = __builtin_amdgcn_wmma_f32_16x16x32_bf16(false, af1.v, false, bf1.v,
                                                      (short)0, acc11, false, false);
    }
  }

  // D layout: lane holds (m = g + 8*half, n = lr) -> 8 consecutive l, fixed co.
  const float bco0 = bias[lr];
  const float bco1 = bias[16 + lr];
#define STORE_ACC(ACC, TM, CT, BCO)                                                  \
  {                                                                                  \
    float* op = out + ((size_t)(b * COUT_ + (CT) * 16 + lr)) * L_ + l0 +             \
                (w * 2 + (TM)) * 16 + 8 * half;                                      \
    v4f lo = { ACC[0] + (BCO), ACC[1] + (BCO), ACC[2] + (BCO), ACC[3] + (BCO) };     \
    v4f hi = { ACC[4] + (BCO), ACC[5] + (BCO), ACC[6] + (BCO), ACC[7] + (BCO) };     \
    *(v4f*)op = lo; *(v4f*)(op + 4) = hi;                                            \
  }
  STORE_ACC(acc00, 0, 0, bco0)
  STORE_ACC(acc01, 0, 1, bco1)
  STORE_ACC(acc10, 1, 0, bco0)
  STORE_ACC(acc11, 1, 1, bco1)
#undef STORE_ACC
}

extern "C" void kernel_launch(void* const* d_in, const int* in_sizes, int n_in,
                              void* d_out, int out_size, void* d_ws, size_t ws_size,
                              hipStream_t stream) {
  (void)in_sizes; (void)n_in; (void)out_size; (void)ws_size;
  const float* x    = (const float*)d_in[0];
  const float* w1   = (const float*)d_in[1];
  const float* b1   = (const float*)d_in[2];
  const float* om1  = (const float*)d_in[3];
  const float* w2   = (const float*)d_in[4];
  const float* b2   = (const float*)d_in[5];
  const float* om2  = (const float*)d_in[6];
  const float* w3   = (const float*)d_in[7];
  const float* b3   = (const float*)d_in[8];
  const float* bias = (const float*)d_in[9];
  float* out = (float*)d_out;
  unsigned short* kbf = (unsigned short*)d_ws;  // 32*32*2080*2 = 4.26 MB scratch

  ck_siren<<<KP_, 32, 0, stream>>>(w1, b1, om1, w2, b2, om2, w3, b3, kbf);
  ck_conv<<<dim3(L_ / LT_, B_), 256, 0, stream>>>(x, kbf, bias, out);
}
